// NonLocalBlock_6064493822100
// MI455X (gfx1250) — matrix operand order repaired and checked by reading, not compiled, not run on hardware
//
#include <hip/hip_runtime.h>

#define BB 8
#define CC 256
#define NN 4096
#define CI 32

#define ROWPITCH 80                         // 64B row + 16B pad (conflict-free b128)
#define KROWS 32
#define VROWS 256
#define BUFBYTES ((KROWS + VROWS) * ROWPITCH)   // 23040 B per buffer

#define AS1 __attribute__((address_space(1)))
#define AS3 __attribute__((address_space(3)))

typedef __attribute__((ext_vector_type(16))) __bf16 v16bf;
typedef __attribute__((ext_vector_type(8)))  __bf16 v8bf;
typedef __attribute__((ext_vector_type(8)))  float  v8f;
typedef __attribute__((ext_vector_type(4)))  unsigned v4u;
typedef int vsi4 __attribute__((vector_size(16)));   // matches async builtin param

union AF { v16bf v; v8bf h[2]; unsigned d[8]; };

#if defined(__has_builtin)
#  if __has_builtin(__builtin_amdgcn_global_load_async_to_lds_b128)
#    define HAVE_ASYNC_LDS 1
#  endif
#endif

__device__ __forceinline__ unsigned short f32_to_bf16(float f) {
    union { float f; unsigned u; } c; c.f = f;
    unsigned u = c.u;
    u += 0x7FFFu + ((u >> 16) & 1u);        // round-to-nearest-even
    return (unsigned short)(u >> 16);
}
__device__ __forceinline__ unsigned pack_bf16(float lo, float hi) {
    return (unsigned)f32_to_bf16(lo) | ((unsigned)f32_to_bf16(hi) << 16);
}

// ---------------------------------------------------------------------------
// Weight conversion to bf16 (row-major = fragment feed order). 81920 elems.
// ---------------------------------------------------------------------------
__global__ void conv_weights_kernel(const float* __restrict__ Wq,
                                    const float* __restrict__ Wk,
                                    const float* __restrict__ Wv,
                                    unsigned short* __restrict__ wqb,
                                    unsigned short* __restrict__ wkb,
                                    unsigned short* __restrict__ wvb) {
    const int i = blockIdx.x * 256 + threadIdx.x;
    const int NQ = CI * CC;                 // 8192
    if (i < NQ)            wqb[i]          = f32_to_bf16(Wq[i]);
    else if (i < 2 * NQ)   wkb[i - NQ]     = f32_to_bf16(Wk[i - NQ]);
    else if (i - 2 * NQ < CC * CC) {
        const int j = i - 2 * NQ;
        wvb[j] = f32_to_bf16(Wv[j]);
    }
}

// ---------------------------------------------------------------------------
// LDS-tiled transpose + bf16 convert: x [B][C][N] f32 -> xT [B][N][C] bf16.
// Coalesced on both sides; pure bandwidth (~48MB total).
// ---------------------------------------------------------------------------
__global__ __launch_bounds__(256) void xT_kernel(const float* __restrict__ x,
                                                 unsigned short* __restrict__ xT) {
    __shared__ float tile[32][33];          // +1 pad: conflict-free transpose
    const int tx = threadIdx.x;             // 0..31
    const int ty = threadIdx.y;             // 0..7
    const int b  = blockIdx.z;
    const int n0 = blockIdx.x * 32;
    const int c0 = blockIdx.y * 32;
    const float* xb = x + (size_t)b * CC * NN;
#pragma unroll
    for (int j = 0; j < 4; ++j) {
        const int cl = ty + j * 8;
        tile[cl][tx] = xb[(size_t)(c0 + cl) * NN + n0 + tx];
    }
    __syncthreads();
    unsigned short* xo = xT + (size_t)b * NN * CC;
#pragma unroll
    for (int j = 0; j < 4; ++j) {
        const int nl = ty + j * 8;
        xo[(size_t)(n0 + nl) * CC + c0 + tx] = f32_to_bf16(tile[tx][nl]);
    }
}

// ---------------------------------------------------------------------------
// WMMA projections. One wave per 16-pixel tile; K-loop over C=256 in 8 steps.
//   v[c,n]  : D(M=c, N=n) = Wv_tile(16c x 32k) @ x(32k x 16n)      16 wmma/step
//   q[n,o]  : D(M=n, N=o) = xT_tile(16n x 32k) @ Wq^T(32k x 16o)    2 wmma/step
//   kT[n,o] : same with Wk                                          2 wmma/step
// ---------------------------------------------------------------------------
__global__ __launch_bounds__(128) void qkv_wmma_kernel(
        const unsigned short* __restrict__ xT,   // [B][N][C] bf16
        const unsigned short* __restrict__ Wqb,  // [CI][C]
        const unsigned short* __restrict__ Wkb,  // [CI][C]
        const unsigned short* __restrict__ Wvb,  // [C][C]
        const float* __restrict__ bq, const float* __restrict__ bk,
        const float* __restrict__ bv,
        unsigned short* __restrict__ q,          // [B][N][CI]
        unsigned short* __restrict__ kT,         // [B][N][CI]
        unsigned short* __restrict__ v) {        // [B][C][N]
    const int tid  = threadIdx.x;
    const int wave = tid >> 5;
    const int lane = tid & 31;
    const int l    = lane & 15;
    const int hi   = lane >> 4;
    const int b    = blockIdx.y;
    const int n0   = blockIdx.x * 64 + wave * 16;

    const unsigned short* xb = xT + (size_t)b * NN * CC;

    const v8f zacc = {0.f, 0.f, 0.f, 0.f, 0.f, 0.f, 0.f, 0.f};
    v8f accV[16], accQ[2], accK[2];
#pragma unroll
    for (int t = 0; t < 16; ++t) accV[t] = zacc;
    accQ[0] = accQ[1] = zacc;
    accK[0] = accK[1] = zacc;

    for (int kb = 0; kb < CC; kb += 32) {
        // x as B-frag: lane l -> col n0+l; elems K = {kb+hi*8+e, kb+16+hi*8+e}
        AF xbf;
        const unsigned short* xr = xb + (size_t)(n0 + l) * CC + kb + hi * 8;
        xbf.h[0] = *(const v8bf*)xr;  xbf.h[1] = *(const v8bf*)(xr + 16);
        // x as A-frag: lane l -> row n0+l; elems K = kb + hi*16 + e (32B chunk)
        AF xaf;
        const unsigned short* xr2 = xb + (size_t)(n0 + l) * CC + kb;
        xaf.h[0] = *(const v8bf*)(xr2 + hi * 16);
        xaf.h[1] = *(const v8bf*)(xr2 + hi * 16 + 8);

        // ---- V: 16 channel tiles
#pragma unroll
        for (int t = 0; t < 16; ++t) {
            const unsigned short* wr = Wvb + (size_t)(t * 16 + l) * CC + kb;
            AF wf;                            // A-frag: row c = t*16+l
            wf.h[0] = *(const v8bf*)(wr + hi * 8);
            wf.h[1] = *(const v8bf*)(wr + 16 + hi * 8);
            accV[t] = __builtin_amdgcn_wmma_f32_16x16x32_bf16(false, wf.v, false, xbf.v,
                                                              (short)0, accV[t], false, false);
        }
        // ---- Q/K: 2 output tiles each (Ci = 32)
#pragma unroll
        for (int ot = 0; ot < 2; ++ot) {
            const unsigned short* wq = Wqb + (size_t)(ot * 16 + l) * CC + kb + hi * 16;
            AF wf;                            // B-frag: col o = ot*16+l
            wf.h[0] = *(const v8bf*)wq;  wf.h[1] = *(const v8bf*)(wq + 8);
            accQ[ot] = __builtin_amdgcn_wmma_f32_16x16x32_bf16(false, xaf.v, false, wf.v,
                                                               (short)0, accQ[ot], false, false);
            const unsigned short* wk = Wkb + (size_t)(ot * 16 + l) * CC + kb + hi * 16;
            AF wf2;
            wf2.h[0] = *(const v8bf*)wk;  wf2.h[1] = *(const v8bf*)(wk + 8);
            accK[ot] = __builtin_amdgcn_wmma_f32_16x16x32_bf16(false, xaf.v, false, wf2.v,
                                                               (short)0, accK[ot], false, false);
        }
    }

    // ---- store v: D(M = c = t*16+r+hi*8, N = n = n0+l)
#pragma unroll
    for (int t = 0; t < 16; ++t)
#pragma unroll
        for (int r = 0; r < 8; ++r) {
            const int c = t * 16 + r + hi * 8;
            v[((size_t)b * CC + c) * NN + n0 + l] = f32_to_bf16(accV[t][r] + bv[c]);
        }
    // ---- store q, kT: D(M = n = n0+r+hi*8, N = o = ot*16+l)
#pragma unroll
    for (int ot = 0; ot < 2; ++ot)
#pragma unroll
        for (int r = 0; r < 8; ++r) {
            const int n = n0 + r + hi * 8;
            const int o = ot * 16 + l;
            q [((size_t)b * NN + n) * CI + o] = f32_to_bf16(accQ[ot][r] + bq[o]);
            kT[((size_t)b * NN + n) * CI + o] = f32_to_bf16(accK[ot][r] + bk[o]);
        }
}

// ---------------------------------------------------------------------------
// Cooperative staging of one 32-key block: K (2KB) + V (16KB) into LDS.
// ---------------------------------------------------------------------------
__device__ __forceinline__ void async_b128(const unsigned short* g, char* lp) {
#ifdef HAVE_ASYNC_LDS
    __builtin_amdgcn_global_load_async_to_lds_b128(
        (vsi4*)g,
        (AS3 vsi4*)(unsigned)(unsigned long long)lp, 0, 0);
#else
    *(v4u*)lp = *(const v4u*)g;
#endif
}

__device__ __forceinline__ void stage_block(const unsigned short* __restrict__ kb,
                                            const unsigned short* __restrict__ vb,
                                            char* lb, int nb, int tid) {
    {   // K rows
        const int row = tid >> 2, ch = tid & 3;
        async_b128(kb + (size_t)(nb + row) * CI + ch * 8,
                   lb + row * ROWPITCH + ch * 16);
    }
#pragma unroll
    for (int j = 0; j < 8; ++j) {           // V rows
        const int vc = tid + 128 * j;
        const int c = vc >> 2, ch = vc & 3;
        async_b128(vb + (size_t)c * NN + nb + ch * 8,
                   lb + (KROWS + c) * ROWPITCH + ch * 16);
    }
}

__device__ __forceinline__ void wait_async_done() {
#ifdef HAVE_ASYNC_LDS
#  if __has_builtin(__builtin_amdgcn_s_wait_asynccnt)
    __builtin_amdgcn_s_wait_asynccnt(0);
#  else
    asm volatile("s_wait_asynccnt 0" ::: "memory");
#  endif
#endif
}

// ---------------------------------------------------------------------------
// Online softmax over a 32-key block for one 16-query set + build P^T B-frag.
// ---------------------------------------------------------------------------
__device__ __forceinline__ AF softmax_p(const v8f& st0, const v8f& st1, int hi,
                                        float& mrun, float& lrun, v8f acc[8]) {
    float mx = fmaxf(st0[0], st1[0]);
#pragma unroll
    for (int r = 1; r < 8; ++r) mx = fmaxf(mx, fmaxf(st0[r], st1[r]));
    mx = fmaxf(mx, __shfl_xor(mx, 16, 32));
    const float mnew = fmaxf(mrun, mx);

    float p0[8], p1[8], rs = 0.f;
#pragma unroll
    for (int r = 0; r < 8; ++r) {
        p0[r] = __expf(st0[r] - mnew);
        p1[r] = __expf(st1[r] - mnew);
        rs += p0[r] + p1[r];
    }
    rs += __shfl_xor(rs, 16, 32);

    if (__any(mnew > mrun)) {               // wave-uniform: rescale only if max moved
        const float sc = __expf(mrun - mnew);
#pragma unroll
        for (int t = 0; t < 8; ++t)
#pragma unroll
            for (int r = 0; r < 8; ++r) acc[t][r] *= sc;
        lrun = lrun * sc + rs;
    } else {
        lrun += rs;
    }
    mrun = mnew;

    unsigned a[4], bq[4], rxc[4];
#pragma unroll
    for (int i = 0; i < 4; ++i) {
        a[i]  = pack_bf16(p0[2 * i], p0[2 * i + 1]);   // keys hi*8   + {2i,2i+1}
        bq[i] = pack_bf16(p1[2 * i], p1[2 * i + 1]);   // keys 16+hi*8+ {2i,2i+1}
    }
#pragma unroll
    for (int i = 0; i < 4; ++i)
        rxc[i] = (unsigned)__shfl_xor((int)(hi ? a[i] : bq[i]), 16, 32);
    AF pb;
#pragma unroll
    for (int i = 0; i < 4; ++i) {
        pb.d[i]     = hi ? rxc[i] : a[i];
        pb.d[4 + i] = hi ? bq[i]  : rxc[i];
    }
    return pb;
}

// ---------------------------------------------------------------------------
// Fused flash-attention. Block = 4 waves: 2 query tiles (32 q) x 2 C halves.
// ---------------------------------------------------------------------------
__global__ __launch_bounds__(128) void attn_kernel(
        const unsigned short* __restrict__ q,
        const unsigned short* __restrict__ kT,
        const unsigned short* __restrict__ v,
        const float* __restrict__ x,
        const float* __restrict__ gammap,
        float* __restrict__ out) {
    __shared__ __align__(16) char smem[2 * BUFBYTES];

    const int tid  = threadIdx.x;
    const int wave = tid >> 5;
    const int lane = tid & 31;
    const int l    = lane & 15;
    const int hi   = lane >> 4;
    const int b    = blockIdx.y;
    const int mq0   = blockIdx.x * 64 + (wave >> 1) * 32;
    const int cbase = (wave & 1) * 128;

    AF qb0, qb1;
    {
        const unsigned short* qr = q + ((size_t)b * NN + mq0 + l) * CI + hi * 16;
        qb0.h[0] = *(const v8bf*)qr;  qb0.h[1] = *(const v8bf*)(qr + 8);
    }
    {
        const unsigned short* qr = q + ((size_t)b * NN + mq0 + 16 + l) * CI + hi * 16;
        qb1.h[0] = *(const v8bf*)qr;  qb1.h[1] = *(const v8bf*)(qr + 8);
    }

    const v8f zacc = {0.f, 0.f, 0.f, 0.f, 0.f, 0.f, 0.f, 0.f};
    v8f accA[8], accB[8];
#pragma unroll
    for (int t = 0; t < 8; ++t) { accA[t] = zacc; accB[t] = zacc; }
    float mrun0 = -1e30f, lrun0 = 0.f, mrun1 = -1e30f, lrun1 = 0.f;

    const unsigned short* kb = kT + (size_t)b * NN * CI;
    const unsigned short* vb = v  + (size_t)b * CC * NN;

    stage_block(kb, vb, smem, 0, tid);

    for (int nb = 0, it = 0; nb < NN; nb += 32, ++it) {
        char* cur = smem + (it & 1) * BUFBYTES;
        wait_async_done();
        __syncthreads();
        if (nb + 32 < NN)
            stage_block(kb, vb, smem + ((it + 1) & 1) * BUFBYTES, nb + 32, tid);

        AF k0, k1;
        {
            const char* kp = cur + l * ROWPITCH + hi * 32;
            k0.h[0] = *(const v8bf*)kp;  k0.h[1] = *(const v8bf*)(kp + 16);
        }
        {
            const char* kp = cur + (16 + l) * ROWPITCH + hi * 32;
            k1.h[0] = *(const v8bf*)kp;  k1.h[1] = *(const v8bf*)(kp + 16);
        }

        v8f st0a = __builtin_amdgcn_wmma_f32_16x16x32_bf16(false, k0.v, false, qb0.v,
                                                           (short)0, zacc, false, false);
        v8f st1a = __builtin_amdgcn_wmma_f32_16x16x32_bf16(false, k1.v, false, qb0.v,
                                                           (short)0, zacc, false, false);
        v8f st0b = __builtin_amdgcn_wmma_f32_16x16x32_bf16(false, k0.v, false, qb1.v,
                                                           (short)0, zacc, false, false);
        v8f st1b = __builtin_amdgcn_wmma_f32_16x16x32_bf16(false, k1.v, false, qb1.v,
                                                           (short)0, zacc, false, false);

        AF pb0 = softmax_p(st0a, st1a, hi, mrun0, lrun0, accA);
        AF pb1 = softmax_p(st0b, st1b, hi, mrun1, lrun1, accB);

#pragma unroll
        for (int t = 0; t < 8; ++t) {
            const int c = cbase + t * 16 + l;
            const char* vp = cur + (KROWS + c) * ROWPITCH + hi * 32;
            AF vf;
            vf.h[0] = *(const v8bf*)vp;  vf.h[1] = *(const v8bf*)(vp + 16);
            accA[t] = __builtin_amdgcn_wmma_f32_16x16x32_bf16(false, vf.v, false, pb0.v,
                                                              (short)0, accA[t], false, false);
            accB[t] = __builtin_amdgcn_wmma_f32_16x16x32_bf16(false, vf.v, false, pb1.v,
                                                              (short)0, accB[t], false, false);
        }
    }

    const float g = gammap[0];
    const float inv0 = 1.0f / lrun0;
    const float inv1 = 1.0f / lrun1;
#pragma unroll
    for (int t = 0; t < 8; ++t) {
#pragma unroll
        for (int r = 0; r < 8; ++r) {
            const int c = cbase + t * 16 + r + hi * 8;
            const size_t i0 = ((size_t)b * CC + c) * NN + mq0 + l;
            out[i0] = g * (accA[t][r] * inv0) + x[i0];
            const size_t i1 = i0 + 16;
            out[i1] = g * (accB[t][r] * inv1) + x[i1];
        }
    }
}

// ---------------------------------------------------------------------------
extern "C" void kernel_launch(void* const* d_in, const int* in_sizes, int n_in,
                              void* d_out, int out_size, void* d_ws, size_t ws_size,
                              hipStream_t stream) {
    const float* x     = (const float*)d_in[0];
    const float* Wq    = (const float*)d_in[1];
    const float* bq    = (const float*)d_in[2];
    const float* Wk    = (const float*)d_in[3];
    const float* bk    = (const float*)d_in[4];
    const float* Wv    = (const float*)d_in[5];
    const float* bv    = (const float*)d_in[6];
    const float* gamma = (const float*)d_in[7];
    float* out = (float*)d_out;

    // Workspace (u16 bf16): q | kT | v | xT | Wqb | Wkb | Wvb  (~37.8 MB)
    unsigned short* qw  = (unsigned short*)d_ws;
    unsigned short* kw  = qw  + (size_t)BB * NN * CI;
    unsigned short* vw  = kw  + (size_t)BB * NN * CI;
    unsigned short* xTb = vw  + (size_t)BB * CC * NN;
    unsigned short* wqb = xTb + (size_t)BB * NN * CC;
    unsigned short* wkb = wqb + (size_t)CI * CC;
    unsigned short* wvb = wkb + (size_t)CI * CC;

    conv_weights_kernel<<<(2 * CI * CC + CC * CC + 255) / 256, 256, 0, stream>>>(
        Wq, Wk, Wv, wqb, wkb, wvb);

    dim3 gt(NN / 32, CC / 32, BB);
    xT_kernel<<<gt, dim3(32, 8), 0, stream>>>(x, xTb);

    dim3 gp(NN / 64, BB);
    qkv_wmma_kernel<<<gp, 128, 0, stream>>>(xTb, wqb, wkb, wvb, bq, bk, bv,
                                            qw, kw, vw);

    dim3 ga(NN / 64, BB);
    attn_kernel<<<ga, 128, 0, stream>>>(qw, kw, vw, x, gamma, out);
}
